// GCNNet_46162308497633
// MI455X (gfx1250) — compile-verified, matrix operand order checked by reading, not run on hardware
//
#include <hip/hip_runtime.h>
#include <hip/hip_bf16.h>
#include <hip/hip_fp16.h>

typedef __attribute__((ext_vector_type(16))) _Float16 v16h;
typedef __attribute__((ext_vector_type(8)))  _Float16 v8h;
typedef __attribute__((ext_vector_type(8)))  float    v8f;

#define NNODES 100000
#define NEDGES 1600000
#define NGRAPH 128
#define HID    128
#define NLAYER 4
#define BN_EPS 1e-5f

// ---------------- utility ----------------
__global__ void k_zero4(float4* __restrict__ p, int n4) {
  int i = blockIdx.x * blockDim.x + threadIdx.x;
  if (i < n4) p[i] = make_float4(0.f, 0.f, 0.f, 0.f);
}

// ---------------- degrees ----------------
__global__ void k_degrees(const int* __restrict__ src, const int* __restrict__ dst,
                          float* __restrict__ deg_out, float* __restrict__ deg_in) {
  int e = blockIdx.x * blockDim.x + threadIdx.x;
  if (e < NEDGES) {
    atomicAdd(deg_out + src[e], 1.0f);
    atomicAdd(deg_in  + dst[e], 1.0f);
  }
}

__global__ void k_invsqrt(float* __restrict__ d, int n) {
  int i = blockIdx.x * blockDim.x + threadIdx.x;
  if (i < n) d[i] = rsqrtf(fmaxf(d[i], 1.0f));
}

// ---------------- pack W[128,128] (fp32, row-major) into WMMA B-fragment layout (fp16) ----
// Bp layout: [tile(8)][kstep(4)][lane(32)][16 halves] => 32KB, L2-resident.
__global__ __launch_bounds__(128)
void k_pack_B(const float* __restrict__ W, _Float16* __restrict__ Bp) {
  const int tid  = blockIdx.x * 128 + threadIdx.x;   // 0..1023
  const int frag = tid >> 5;                          // tile*4 + kstep
  const int lane = tid & 31;
  const int tile = frag >> 2;
  const int kt   = frag & 3;
  const int hi   = lane >> 4;
  const int col  = tile * 16 + (lane & 15);
  _Float16* dstp = Bp + ((size_t)frag * 32 + lane) * 16;
#pragma unroll
  for (int j = 0; j < 16; ++j) {
    const int K = kt * 32 + (j & 7) + 8 * hi + ((j >> 3) << 4);
    dstp[j] = (_Float16)W[(size_t)K * HID + col];
  }
}

// ---------------- WMMA GEMM (fp32 A, embed): Ch[M,128] = A @ W + bias, fp16 out -------
__global__ __launch_bounds__(128)
void k_wmma_gemm_f32A(const float* __restrict__ A, const _Float16* __restrict__ Bp,
                      const float* __restrict__ bias, _Float16* __restrict__ Ch) {
  const int lane = threadIdx.x & 31;
  const int wave = threadIdx.x >> 5;
  const int hi   = lane >> 4;
  const int mr   = lane & 15;
  const int row0 = blockIdx.x * 32;

  v16h bfrag[2][4];
#pragma unroll
  for (int t = 0; t < 2; ++t)
#pragma unroll
    for (int kt = 0; kt < 4; ++kt)
      bfrag[t][kt] = *(const v16h*)(Bp + ((size_t)((wave * 2 + t) * 4 + kt) * 32 + lane) * 16);

  const float bias0 = bias[(wave * 2 + 0) * 16 + mr];
  const float bias1 = bias[(wave * 2 + 1) * 16 + mr];

#pragma unroll
  for (int mt = 0; mt < 2; ++mt) {
    const int row = row0 + mt * 16 + mr;
    const float4* arow = (const float4*)(A + (size_t)row * HID);
    v16h afrag[4];
#pragma unroll
    for (int kt = 0; kt < 4; ++kt) {
      const float4 p0 = arow[kt * 8 + 2 * hi];
      const float4 p1 = arow[kt * 8 + 2 * hi + 1];
      const float4 p2 = arow[kt * 8 + 4 + 2 * hi];
      const float4 p3 = arow[kt * 8 + 4 + 2 * hi + 1];
      afrag[kt][0]  = (_Float16)p0.x; afrag[kt][1]  = (_Float16)p0.y;
      afrag[kt][2]  = (_Float16)p0.z; afrag[kt][3]  = (_Float16)p0.w;
      afrag[kt][4]  = (_Float16)p1.x; afrag[kt][5]  = (_Float16)p1.y;
      afrag[kt][6]  = (_Float16)p1.z; afrag[kt][7]  = (_Float16)p1.w;
      afrag[kt][8]  = (_Float16)p2.x; afrag[kt][9]  = (_Float16)p2.y;
      afrag[kt][10] = (_Float16)p2.z; afrag[kt][11] = (_Float16)p2.w;
      afrag[kt][12] = (_Float16)p3.x; afrag[kt][13] = (_Float16)p3.y;
      afrag[kt][14] = (_Float16)p3.z; afrag[kt][15] = (_Float16)p3.w;
    }
#pragma unroll
    for (int t = 0; t < 2; ++t) {
      const int col = (wave * 2 + t) * 16 + mr;
      v8f acc = {};
#pragma unroll
      for (int kt = 0; kt < 4; ++kt)
        acc = __builtin_amdgcn_wmma_f32_16x16x32_f16(
            false, afrag[kt], false, bfrag[t][kt], (short)0, acc, false, false);
      const float bb = t ? bias1 : bias0;
#pragma unroll
      for (int r = 0; r < 8; ++r) {
        const int m = mt * 16 + r + 8 * hi;
        Ch[(size_t)(row0 + m) * HID + col] = (_Float16)(acc[r] + bb);
      }
    }
  }
}

// ---------------- WMMA GEMM (fp16 A + row scale, layers): Ch[M,128] fp16 ----------------
// A fragments load directly as two 16B half-runs per K-step; scale fused as v_pk_mul_f16.
__global__ __launch_bounds__(128)
void k_wmma_gemm_f16A(const _Float16* __restrict__ A, const float* __restrict__ rowScale,
                      const _Float16* __restrict__ Bp, const float* __restrict__ bias,
                      _Float16* __restrict__ Ch) {
  const int lane = threadIdx.x & 31;
  const int wave = threadIdx.x >> 5;
  const int hi   = lane >> 4;
  const int mr   = lane & 15;
  const int row0 = blockIdx.x * 32;

  v16h bfrag[2][4];
#pragma unroll
  for (int t = 0; t < 2; ++t)
#pragma unroll
    for (int kt = 0; kt < 4; ++kt)
      bfrag[t][kt] = *(const v16h*)(Bp + ((size_t)((wave * 2 + t) * 4 + kt) * 32 + lane) * 16);

  const float bias0 = bias[(wave * 2 + 0) * 16 + mr];
  const float bias1 = bias[(wave * 2 + 1) * 16 + mr];

#pragma unroll
  for (int mt = 0; mt < 2; ++mt) {
    const int row = row0 + mt * 16 + mr;
    const _Float16 sh = (_Float16)rowScale[row];
    const _Float16* arow = A + (size_t)row * HID;

    if (mt == 0 && blockIdx.x + 1 < gridDim.x)
      __builtin_prefetch(arow + 32 * HID, 0, 1);   // next block's A rows

    v16h afrag[4];
#pragma unroll
    for (int kt = 0; kt < 4; ++kt) {
      const v8h g0 = *(const v8h*)(arow + kt * 32 + 8 * hi);        // K run 1 (16B)
      const v8h g1 = *(const v8h*)(arow + kt * 32 + 16 + 8 * hi);   // K run 2 (16B)
#pragma unroll
      for (int j = 0; j < 8; ++j) {
        afrag[kt][j]     = g0[j] * sh;
        afrag[kt][j + 8] = g1[j] * sh;
      }
    }
#pragma unroll
    for (int t = 0; t < 2; ++t) {
      const int col = (wave * 2 + t) * 16 + mr;
      v8f acc = {};
#pragma unroll
      for (int kt = 0; kt < 4; ++kt)
        acc = __builtin_amdgcn_wmma_f32_16x16x32_f16(
            false, afrag[kt], false, bfrag[t][kt], (short)0, acc, false, false);
      const float bb = t ? bias1 : bias0;
#pragma unroll
      for (int r = 0; r < 8; ++r) {
        const int m = mt * 16 + r + 8 * hi;
        Ch[(size_t)(row0 + m) * HID + col] = (_Float16)(acc[r] + bb);
      }
    }
  }
}

// ---------------- edge gather / scatter-add in fp16 (the dominant stage) --------------
// 4 edges per 256-thread block; 64 half2 lanes move one contiguous 256B row.
// Scatter uses packed-fp16 atomics (global_atomic_pk_add_f16).
__global__ __launch_bounds__(256)
void k_edge_agg(const __half2* __restrict__ x2, const float* __restrict__ inv_do,
                const int* __restrict__ src, const int* __restrict__ dst,
                __half2* __restrict__ agg2) {
  const int e = blockIdx.x * 4 + (threadIdx.x >> 6);
  const int f = threadIdx.x & 63;
  if (e < NEDGES) {
    const int s = src[e];
    const int d = dst[e];
    const __half2 v = __hmul2(x2[(size_t)s * (HID / 2) + f], __float2half2_rn(inv_do[s]));
    atomicAdd(agg2 + (size_t)d * (HID / 2) + f, v);
  }
}

// ---------------- BatchNorm stats (fp32 accumulation over fp16 hn) ----------------
__global__ __launch_bounds__(64)
void k_bn_partial(const __half2* __restrict__ hn2,
                  float* __restrict__ bnsum, float* __restrict__ bnsq) {
  const int c2 = threadIdx.x;                   // half2 column 0..63
  float sx = 0.f, sy = 0.f, qx = 0.f, qy = 0.f;
  for (int r = blockIdx.x; r < NNODES; r += gridDim.x) {
    const float2 v = __half22float2(hn2[(size_t)r * (HID / 2) + c2]);
    sx += v.x; sy += v.y;
    qx += v.x * v.x; qy += v.y * v.y;
  }
  atomicAdd(bnsum + 2 * c2 + 0, sx);
  atomicAdd(bnsum + 2 * c2 + 1, sy);
  atomicAdd(bnsq  + 2 * c2 + 0, qx);
  atomicAdd(bnsq  + 2 * c2 + 1, qy);
}

__global__ void k_bn_finalize(const float* __restrict__ bnsum, const float* __restrict__ bnsq,
                              float* __restrict__ mean, float* __restrict__ invstd) {
  const int c = threadIdx.x;
  const float mu  = bnsum[c] * (1.0f / (float)NNODES);
  const float var = bnsq[c] * (1.0f / (float)NNODES) - mu * mu;
  mean[c]   = mu;
  invstd[c] = rsqrtf(var + BN_EPS);
}

// x(fp16) += relu(bn(hn fp16)); half2-vectorized
__global__ void k_bn_apply_h2(const __half2* __restrict__ hn2,
                              const float* __restrict__ mean, const float* __restrict__ invstd,
                              const float* __restrict__ gamma, const float* __restrict__ beta,
                              __half2* __restrict__ x2, int total2) {
  const int i = blockIdx.x * blockDim.x + threadIdx.x;
  if (i >= total2) return;
  const int c0 = (i * 2) & (HID - 1);
  const float2 hv = __half22float2(hn2[i]);
  const float2 xv = __half22float2(x2[i]);
  const float a = xv.x + fmaxf((hv.x - mean[c0 + 0]) * invstd[c0 + 0] * gamma[c0 + 0] + beta[c0 + 0], 0.f);
  const float b = xv.y + fmaxf((hv.y - mean[c0 + 1]) * invstd[c0 + 1] * gamma[c0 + 1] + beta[c0 + 1], 0.f);
  x2[i] = __floats2half2_rn(a, b);
}

// ---------------- graph mean readout (fp32 accumulation) ----------------
__global__ __launch_bounds__(128)
void k_readout_sum(const _Float16* __restrict__ x, const int* __restrict__ gid,
                   float* __restrict__ hg) {
  const int ni = blockIdx.x;
  const int g  = gid[ni];
  atomicAdd(hg + (size_t)g * HID + threadIdx.x, (float)x[(size_t)ni * HID + threadIdx.x]);
}

__global__ void k_count(const int* __restrict__ gid, float* __restrict__ cnt) {
  const int i = blockIdx.x * blockDim.x + threadIdx.x;
  if (i < NNODES) atomicAdd(cnt + gid[i], 1.0f);
}

__global__ __launch_bounds__(128)
void k_readout_div(float* __restrict__ hg, const float* __restrict__ cnt) {
  const int g = blockIdx.x;
  hg[(size_t)g * HID + threadIdx.x] /= fmaxf(cnt[g], 1.0f);
}

// ---------------- tiny MLP readout (fp32, negligible cost) ----------------
__global__ void k_mlp(const float* __restrict__ in, const float* __restrict__ W,
                      const float* __restrict__ b, float* __restrict__ out,
                      int M, int K, int Nc, int relu) {
  const int i = blockIdx.x * blockDim.x + threadIdx.x;
  if (i >= M * Nc) return;
  const int m = i / Nc, n = i % Nc;
  float acc = b[n];
  for (int k = 0; k < K; ++k) acc += in[m * K + k] * W[k * Nc + n];
  out[i] = relu ? fmaxf(acc, 0.0f) : acc;
}

// ---------------- launcher ----------------
extern "C" void kernel_launch(void* const* d_in, const int* in_sizes, int n_in,
                              void* d_out, int out_size, void* d_ws, size_t ws_size,
                              hipStream_t stream) {
  (void)in_sizes; (void)n_in; (void)out_size; (void)ws_size;

  const float* h       = (const float*)d_in[0];   // [N,128]
  const int*   src     = (const int*)  d_in[2];   // [E]
  const int*   dst     = (const int*)  d_in[3];   // [E]
  const int*   gid     = (const int*)  d_in[4];   // [N]
  const float* W_embed = (const float*)d_in[5];   // [128,128]
  const float* b_embed = (const float*)d_in[6];   // [128]
  const float* Wl      = (const float*)d_in[7];   // [4,128,128]
  const float* bl      = (const float*)d_in[8];   // [4,128]
  const float* gamma   = (const float*)d_in[9];   // [4,128]
  const float* beta    = (const float*)d_in[10];  // [4,128]
  const float* W1      = (const float*)d_in[11];  // [128,64]
  const float* b1      = (const float*)d_in[12];  // [64]
  const float* W2      = (const float*)d_in[13];  // [64,32]
  const float* b2      = (const float*)d_in[14];  // [32]
  const float* W3      = (const float*)d_in[15];  // [32,10]
  const float* b3      = (const float*)d_in[16];  // [10]
  float* out = (float*)d_out;                     // [128,10]

  // workspace layout: three fp16 node buffers, then fp32 tail
  float* ws = (float*)d_ws;
  const size_t NF = (size_t)NNODES * HID;
  _Float16* xh   = (_Float16*)ws;               // [N,128] fp16
  _Float16* aggh = xh + NF;                     // [N,128] fp16
  _Float16* hnh  = aggh + NF;                   // [N,128] fp16 (pre-BN GEMM output)
  float* deg    = (float*)(hnh + NF);           // [2N]
  float* bnsum  = deg    + 2 * NNODES;          // [128]
  float* bnsq   = bnsum  + HID;
  float* mean   = bnsq   + HID;
  float* invstd = mean   + HID;
  float* hg     = invstd + HID;                 // [G,128]
  float* cnt    = hg     + (size_t)NGRAPH * HID;// [G]
  float* hg1    = cnt    + NGRAPH;              // [G,64]
  float* hg2    = hg1    + (size_t)NGRAPH * 64; // [G,32]
  _Float16* Bp  = (_Float16*)(hg2 + (size_t)NGRAPH * 32); // 32KB packed weights
  float* inv_do = deg;
  float* inv_di = deg + NNODES;

  const int ZB = 256;
  // degrees -> inverse sqrt norms
  k_zero4<<<(2 * NNODES / 4 + ZB - 1) / ZB, ZB, 0, stream>>>((float4*)deg, 2 * NNODES / 4);
  k_degrees<<<(NEDGES + ZB - 1) / ZB, ZB, 0, stream>>>(src, dst, inv_do, inv_di);
  k_invsqrt<<<(2 * NNODES + ZB - 1) / ZB, ZB, 0, stream>>>(deg, 2 * NNODES);

  // x(fp16) = h @ W_embed + b_embed
  k_pack_B<<<8, 128, 0, stream>>>(W_embed, Bp);
  k_wmma_gemm_f32A<<<NNODES / 32, 128, 0, stream>>>(h, Bp, b_embed, xh);

  // GCN layers
  for (int l = 0; l < NLAYER; ++l) {
    // zero fp16 agg (NF halves = NF/8 float4)
    k_zero4<<<((int)(NF / 8) + ZB - 1) / ZB, ZB, 0, stream>>>((float4*)aggh, (int)(NF / 8));
    k_edge_agg<<<NEDGES / 4, 256, 0, stream>>>((const __half2*)xh, inv_do, src, dst,
                                               (__half2*)aggh);
    // hn(fp16) = (inv_di * agg) @ Wl[l] + bl[l]
    k_pack_B<<<8, 128, 0, stream>>>(Wl + (size_t)l * HID * HID, Bp);
    k_wmma_gemm_f16A<<<NNODES / 32, 128, 0, stream>>>(aggh, inv_di, Bp,
                                                      bl + (size_t)l * HID, hnh);
    k_zero4<<<1, 64, 0, stream>>>((float4*)bnsum, 64);           // bnsum + bnsq
    k_bn_partial<<<256, 64, 0, stream>>>((const __half2*)hnh, bnsum, bnsq);
    k_bn_finalize<<<1, 128, 0, stream>>>(bnsum, bnsq, mean, invstd);
    k_bn_apply_h2<<<((int)(NF / 2) + ZB - 1) / ZB, ZB, 0, stream>>>(
        (const __half2*)hnh, mean, invstd,
        gamma + (size_t)l * HID, beta + (size_t)l * HID,
        (__half2*)xh, (int)(NF / 2));
  }

  // per-graph mean readout
  k_zero4<<<((NGRAPH * HID + NGRAPH) / 4 + ZB - 1) / ZB, ZB, 0, stream>>>(
      (float4*)hg, (NGRAPH * HID + NGRAPH) / 4);
  k_readout_sum<<<NNODES, 128, 0, stream>>>(xh, gid, hg);
  k_count<<<(NNODES + ZB - 1) / ZB, ZB, 0, stream>>>(gid, cnt);
  k_readout_div<<<NGRAPH, 128, 0, stream>>>(hg, cnt);

  // MLP readout
  k_mlp<<<(NGRAPH * 64 + 127) / 128, 128, 0, stream>>>(hg,  W1, b1, hg1, NGRAPH, 128, 64, 1);
  k_mlp<<<(NGRAPH * 32 + 127) / 128, 128, 0, stream>>>(hg1, W2, b2, hg2, NGRAPH, 64, 32, 1);
  k_mlp<<<(NGRAPH * 10 + 127) / 128, 128, 0, stream>>>(hg2, W3, b3, out, NGRAPH, 32, 10, 0);
}
